// MultiSpanAllocator_592705486931
// MI455X (gfx1250) — matrix operand order repaired and checked by reading, not compiled, not run on hardware
//
#include <hip/hip_runtime.h>
#include <hip/hip_bf16.h>

typedef __attribute__((ext_vector_type(16))) _Float16 v16h;
typedef __attribute__((ext_vector_type(8)))  _Float16 v8h;
typedef __attribute__((ext_vector_type(8)))  float    v8f;

#define S_TOT     4096
#define DHEAD     64
#define NHEAD     8
#define SPANLEN   512
#define RADIUS_SQ 6.25f
#define SCALE     0.125f      // 1/sqrt(64)
#define BLOCK_M   128         // Q rows per 256-thread block
#define WAVE_M    16          // Q rows per wave (one WMMA M tile)
#define KV_TILE   32          // keys per staged tile (one WMMA K for PV)
#define NWAVES    8

// LDS strides in halfs (16B-aligned rows, padded to dodge bank conflicts)
#define KS_STRIDE 72          // K tile: [32 keys][64 d]
#define VT_STRIDE 40          // V tile transposed: [64 d][32 keys]

#define WMMA_F16(A, B, C) \
  __builtin_amdgcn_wmma_f32_16x16x32_f16(false, (A), false, (B), (short)0, (C), false, false)

// ds_swizzle group-of-32 xor mode: offset = (xor_mask<<10) | (0<<5) | and_mask(0x1f)
#define SWZ_XOR(m) (((m) << 10) | 0x1f)
#define RED_MAX(x, imm) \
  (x) = fmaxf((x), __int_as_float(__builtin_amdgcn_ds_swizzle(__float_as_int(x), (imm))))
#define RED_ADD(x, imm) \
  (x) += __int_as_float(__builtin_amdgcn_ds_swizzle(__float_as_int(x), (imm)))

// 16 contiguous halfs from LDS (two ds_load_b128)
static __device__ inline v16h lds_load16(const _Float16* p) {
  v8h lo = *(const v8h*)(p);
  v8h hi = *(const v8h*)(p + 8);
  v16h r;
#pragma unroll
  for (int i = 0; i < 8; ++i) { r[i] = lo[i]; r[8 + i] = hi[i]; }
  return r;
}

// cooperative stage of one 32-key tile: K row-major f16, V transposed f16
static __device__ inline void stage_tile(const float* __restrict__ kh,
                                         const float* __restrict__ vh,
                                         int kbase, _Float16* __restrict__ ksb,
                                         _Float16* __restrict__ vtb, int tid) {
#pragma unroll
  for (int i = 0; i < 8; ++i) {
    const int idx = tid + i * 256;           // 2048 elements = 32 keys x 64 d
    const int key = idx >> 6;
    const int d   = idx & 63;
    const size_t g = (size_t)(kbase + key) * DHEAD + d;
    ksb[key * KS_STRIDE + d] = (_Float16)kh[g];
    vtb[d * VT_STRIDE + key] = (_Float16)vh[g];
  }
}

__global__ __launch_bounds__(256)
void mspan_attn_kernel(const float* __restrict__ q, const float* __restrict__ k,
                       const float* __restrict__ v, const float* __restrict__ coords,
                       const unsigned char* __restrict__ is_causal,
                       float* __restrict__ out) {
  // double-buffered K/V tiles + wave-private P scratch
  __shared__ __attribute__((aligned(16))) _Float16 ks[2][KV_TILE * KS_STRIDE];
  __shared__ __attribute__((aligned(16))) _Float16 vt[2][DHEAD * VT_STRIDE];
  __shared__ __attribute__((aligned(16))) _Float16 pbuf[NWAVES * WAVE_M * KV_TILE];

  const int tid  = threadIdx.x;
  const int wave = tid >> 5;
  const int lane = tid & 31;
  const int hw   = lane >> 4;   // lane-half (0/1) selects K sub-blocks in WMMA layouts
  const int l16  = lane & 15;   // M (A rows) / N (B,C,D columns) position

  const int h      = blockIdx.y;
  const int qblock = blockIdx.x * BLOCK_M;
  const int qwave  = qblock + wave * WAVE_M;
  const int qspan  = qblock / SPANLEN;              // whole block is inside one span
  const bool causal = is_causal[qspan] != 0;

  const float* qh = q + (size_t)h * S_TOT * DHEAD;
  const float* kh = k + (size_t)h * S_TOT * DHEAD;
  const float* vh = v + (size_t)h * S_TOT * DHEAD;

  // ---- preload this wave's Q tile into A-matrix registers (f32 -> f16) ----
  // A 16x32 f16 layout: row = l16; halfs h0..15 : K = hw*16 + e
  const float* qrow = qh + (size_t)(qwave + l16) * DHEAD;
  v16h qa0, qa1;   // d chunks [0,32) and [32,64)
#pragma unroll
  for (int e = 0; e < 8; ++e) {
    qa0[e]     = (_Float16)qrow[ 0 + hw * 16 + e];
    qa0[8 + e] = (_Float16)qrow[ 8 + hw * 16 + e];
    qa1[e]     = (_Float16)qrow[32 + hw * 16 + e];
    qa1[8 + e] = (_Float16)qrow[40 + hw * 16 + e];
  }

  // per-lane row metadata: row r = vv + 8*hw of the C/D layout
  int   qi_r[8];
  float qcy[8], qcx[8];
#pragma unroll
  for (int vv = 0; vv < 8; ++vv) {
    const int qi = qwave + vv + 8 * hw;
    qi_r[vv] = qi;
    qcy[vv] = coords[2 * qi];
    qcx[vv] = coords[2 * qi + 1];
  }

  // flash-attention running state; lrow is a PER-LANE partial (reduced only at the end,
  // valid because alpha is row-uniform so all lane partials scale identically)
  float mrow[8], lrow[8];
  v8f o0 = {}, o1 = {}, o2 = {}, o3 = {};   // O columns [0,16) [16,32) [32,48) [48,64)
#pragma unroll
  for (int vv = 0; vv < 8; ++vv) { mrow[vv] = -1e30f; lrow[vv] = 0.0f; }

  // span structure: keys only from spans <= qspan; causal span needs keys < qblock+128 only
  const int nkv    = causal ? (qblock + BLOCK_M) : ((qspan + 1) * SPANLEN);
  const int ntiles = nkv / KV_TILE;
  _Float16* pw = &pbuf[wave * WAVE_M * KV_TILE];  // wave-private P scratch [16][32]

  // prologue: stage tile 0 into buffer 0
  stage_tile(kh, vh, 0, ks[0], vt[0], tid);
  __syncthreads();

  for (int t = 0; t < ntiles; ++t) {
    const int kbase = t * KV_TILE;
    const _Float16* ksb = ks[t & 1];
    const _Float16* vtb = vt[t & 1];

    // stage next tile into the other buffer (overlaps with this tile's WMMAs)
    if (t + 1 < ntiles)
      stage_tile(kh, vh, kbase + KV_TILE, ks[(t + 1) & 1], vt[(t + 1) & 1], tid);
    // cacheline-granular prefetch of tile t+2 (gfx1250 global_prefetch_b8)
    if (t + 2 < ntiles) {
      const size_t pb = (size_t)(kbase + 2 * KV_TILE) * DHEAD;
      if (tid < 64)       __builtin_prefetch(&kh[pb + (size_t)tid * 32], 0, 1);
      else if (tid < 128) __builtin_prefetch(&vh[pb + (size_t)(tid - 64) * 32], 0, 1);
    }

    const int  kspan = kbase >> 9;             // kbase / SPANLEN
    const bool same  = (kspan == qspan);
    // wave-uniform skip: causal diag tiles entirely above this wave's rows
    const bool wave_skip = same && causal && (kbase >= qwave + WAVE_M);

    if (!wave_skip) {
      // ---- S = Q K^T : two 16x16 score tiles (keys kbase+[0,16) and +[16,32)) ----
      // B 32x16 f16 layout: col = l16 (key); halfs h0..15 : K(d) = hw*16 + e
      v8f s0 = {}, s1 = {};
      {
        const _Float16* kr0 = &ksb[(0 * 16 + l16) * KS_STRIDE];
        const _Float16* kr1 = &ksb[(1 * 16 + l16) * KS_STRIDE];
        v16h b;
        b = lds_load16(kr0 +  0 + hw * 16); s0 = WMMA_F16(qa0, b, s0);
        b = lds_load16(kr0 + 32 + hw * 16); s0 = WMMA_F16(qa1, b, s0);
        b = lds_load16(kr1 +  0 + hw * 16); s1 = WMMA_F16(qa0, b, s1);
        b = lds_load16(kr1 + 32 + hw * 16); s1 = WMMA_F16(qa1, b, s1);
      }

      // ---- mask + online softmax (C/D layout: row = vv+8*hw, col = l16) ----
      const int kj0 = kbase + l16;
      const int kj1 = kj0 + 16;
      float kcy0 = 0.f, kcx0 = 0.f, kcy1 = 0.f, kcx1 = 0.f;
      if (same) {
        kcy0 = coords[2 * kj0]; kcx0 = coords[2 * kj0 + 1];
        kcy1 = coords[2 * kj1]; kcx1 = coords[2 * kj1 + 1];
      }

      float alpha[8];
#pragma unroll
      for (int vv = 0; vv < 8; ++vv) {
        float x0 = s0[vv] * SCALE;
        float x1 = s1[vv] * SCALE;
        if (same) {   // history spans are fully valid; same-span needs time & space test
          const float dy0 = qcy[vv] - kcy0, dx0 = qcx[vv] - kcx0;
          const float dy1 = qcy[vv] - kcy1, dx1 = qcx[vv] - kcx1;
          const bool ok0 = (!causal || (qi_r[vv] >= kj0)) && (dy0 * dy0 + dx0 * dx0 < RADIUS_SQ);
          const bool ok1 = (!causal || (qi_r[vv] >= kj1)) && (dy1 * dy1 + dx1 * dx1 < RADIUS_SQ);
          x0 = ok0 ? x0 : -1e30f;
          x1 = ok1 ? x1 : -1e30f;
        }
        // row max across the 16 lanes of this lane-half (xor stays inside the half)
        float rm = fmaxf(x0, x1);
        RED_MAX(rm, SWZ_XOR(1));
        RED_MAX(rm, SWZ_XOR(2));
        RED_MAX(rm, SWZ_XOR(4));
        RED_MAX(rm, SWZ_XOR(8));
        const float mn = fmaxf(mrow[vv], rm);
        const float al = __expf(mrow[vv] - mn);   // row-uniform
        mrow[vv] = mn;
        const float p0 = __expf(x0 - mn);
        const float p1 = __expf(x1 - mn);
        lrow[vv] = lrow[vv] * al + (p0 + p1);     // per-lane partial, no per-tile reduce
        alpha[vv] = al;
        // spill P (f16) to wave-private LDS in [row][key] order for A-relayout
        const int r = vv + 8 * hw;
        pw[r * KV_TILE + l16]      = (_Float16)p0;
        pw[r * KV_TILE + 16 + l16] = (_Float16)p1;
      }
#pragma unroll
      for (int vv = 0; vv < 8; ++vv) {
        o0[vv] *= alpha[vv]; o1[vv] *= alpha[vv];
        o2[vv] *= alpha[vv]; o3[vv] *= alpha[vv];
      }

      // wave-internal LDS ordering: P stores -> A-layout loads
      asm volatile("s_wait_dscnt 0" ::: "memory");

      // P as A-matrix 16x32: row = l16; h0..15: kk = hw*16+e
      v16h pa;
      {
        const _Float16* prow = pw + l16 * KV_TILE;
        v8h plo = *(const v8h*)(prow + hw * 16);
        v8h phi = *(const v8h*)(prow + 8 + hw * 16);
#pragma unroll
        for (int i = 0; i < 8; ++i) { pa[i] = plo[i]; pa[8 + i] = phi[i]; }
      }

      // ---- O += P V : B 32x16 f16: col = d = t*16+l16; halfs: kk = hw*16 + e ----
      {
        v16h vb;
        vb = lds_load16(&vtb[(0 * 16 + l16) * VT_STRIDE + hw * 16]); o0 = WMMA_F16(pa, vb, o0);
        vb = lds_load16(&vtb[(1 * 16 + l16) * VT_STRIDE + hw * 16]); o1 = WMMA_F16(pa, vb, o1);
        vb = lds_load16(&vtb[(2 * 16 + l16) * VT_STRIDE + hw * 16]); o2 = WMMA_F16(pa, vb, o2);
        vb = lds_load16(&vtb[(3 * 16 + l16) * VT_STRIDE + hw * 16]); o3 = WMMA_F16(pa, vb, o3);
      }
    }
    // single barrier per iteration: orders next-tile staging (done above) against next
    // iteration's compute, and this iteration's compute against re-staging of this buffer
    __syncthreads();
  }

  // ---- epilogue: reduce per-lane l partials once, then out = O / l ----
  float* ob = out + (size_t)h * S_TOT * DHEAD;
#pragma unroll
  for (int vv = 0; vv < 8; ++vv) {
    float rs = lrow[vv];
    RED_ADD(rs, SWZ_XOR(1));
    RED_ADD(rs, SWZ_XOR(2));
    RED_ADD(rs, SWZ_XOR(4));
    RED_ADD(rs, SWZ_XOR(8));
    const float inv = 1.0f / rs;
    float* orow = ob + (size_t)qi_r[vv] * DHEAD;
    orow[ 0 + l16] = o0[vv] * inv;
    orow[16 + l16] = o1[vv] * inv;
    orow[32 + l16] = o2[vv] * inv;
    orow[48 + l16] = o3[vv] * inv;
  }
}

extern "C" void kernel_launch(void* const* d_in, const int* in_sizes, int n_in,
                              void* d_out, int out_size, void* d_ws, size_t ws_size,
                              hipStream_t stream) {
  (void)in_sizes; (void)n_in; (void)out_size; (void)d_ws; (void)ws_size;
  const float* q      = (const float*)d_in[0];
  const float* k      = (const float*)d_in[1];
  const float* v      = (const float*)d_in[2];
  const float* coords = (const float*)d_in[3];
  (void)d_in[4];  // span_ids: derived as key>>9 on device
  const unsigned char* is_causal = (const unsigned char*)d_in[5];
  float* out = (float*)d_out;

  dim3 grid(S_TOT / BLOCK_M, NHEAD);   // (32, 8)
  mspan_attn_kernel<<<grid, 256, 0, stream>>>(q, k, v, coords, is_causal, out);
}
